// GRUArbiter_18846316495057
// MI455X (gfx1250) — matrix operand order repaired and checked by reading, not compiled, not run on hardware
//
#include <hip/hip_runtime.h>
#include <hip/hip_bf16.h>

typedef __attribute__((ext_vector_type(16))) __bf16 v16bf;
typedef __attribute__((ext_vector_type(8)))  float  v8f;

#define Bb 8
#define Ll 2048
#define Dd 1024
#define K1 2048   // 2*D
#define N1 3072   // 3*D
#define M1 16384  // B*L
#define SCAN_WGS 64

union Frag16 {
  uint4  q[2];
  v16bf  v;
};

__device__ __forceinline__ float sigmoidf_(float x) { return 1.0f / (1.0f + __expf(-x)); }

// ------------------------------------------------------------------
// init: f32 -> bf16 weight conversion, h buffers, logits, barrier
// ------------------------------------------------------------------
__global__ void init_ws_kernel(const float* __restrict__ w_ih,
                               const float* __restrict__ w_hh,
                               const float* __restrict__ hidden,
                               void* wih_bf_, void* whh_bf_, void* hbuf_,
                               float* __restrict__ logits,
                               unsigned* __restrict__ bar) {
  __bf16* Wih  = (__bf16*)wih_bf_;
  __bf16* Whh  = (__bf16*)whh_bf_;
  __bf16* hbuf = (__bf16*)hbuf_;  // [2][8][Dd]
  const size_t n_wih = (size_t)N1 * K1;
  const size_t n_whh = (size_t)N1 * Dd;
  const size_t n_h   = (size_t)2 * Bb * Dd;
  const size_t n_lg  = (size_t)Ll * 16;
  const size_t total = n_wih + n_whh + n_h + n_lg + 2;
  for (size_t i = (size_t)blockIdx.x * blockDim.x + threadIdx.x; i < total;
       i += (size_t)gridDim.x * blockDim.x) {
    if (i < n_wih) {
      Wih[i] = (__bf16)w_ih[i];
    } else if (i < n_wih + n_whh) {
      size_t j = i - n_wih;
      Whh[j] = (__bf16)w_hh[j];
    } else if (i < n_wih + n_whh + n_h) {
      size_t j = i - n_wih - n_whh;
      int buf = (int)(j / (Bb * Dd));
      int row = (int)((j % (Bb * Dd)) / Dd);
      int dd  = (int)(j % Dd);
      float v = (buf == 0) ? hidden[row * Dd + dd] : 0.0f;
      hbuf[j] = (__bf16)v;
    } else if (i < n_wih + n_whh + n_h + n_lg) {
      logits[i - n_wih - n_whh - n_h] = 0.0f;
    } else {
      bar[i - n_wih - n_whh - n_h - n_lg] = 0u;
    }
  }
}

// ------------------------------------------------------------------
// Phase 1: gi = [rwkv|mamba] @ W_ih^T + b_ih  (16384x3072, K=2048)
// One wave computes a 32x64 tile (2 M-tiles x 4 N-tiles, 8 accums);
// B fragments are shared across the two M-tiles (halves W_ih traffic).
// ------------------------------------------------------------------
__global__ __launch_bounds__(256) void gemm_ih_kernel(
    const float* __restrict__ rwkv, const float* __restrict__ mamba,
    const void* __restrict__ wih_bf_, const float* __restrict__ b_ih,
    float* __restrict__ gi) {
  const __bf16* Wih = (const __bf16*)wih_bf_;
  const int wave = threadIdx.x >> 5;
  const int lane = threadIdx.x & 31;
  const int l15  = lane & 15;
  const int g    = lane >> 4;

  const int idx   = blockIdx.x * 8 + wave;       // 0 .. 24575
  const int tileM = (idx / 48) * 32;             // 512 M-pair tiles
  const int tileN = (idx % 48) * 64;             // 48 N64 tiles

  const int m0 = tileM + l15;
  const int m1 = m0 + 16;
  const float* rw0 = rwkv  + (size_t)m0 * Dd;
  const float* mb0 = mamba + (size_t)m0 * Dd;
  const float* rw1 = rwkv  + (size_t)m1 * Dd;
  const float* mb1 = mamba + (size_t)m1 * Dd;

  v8f acc00 = {}, acc01 = {}, acc02 = {}, acc03 = {};
  v8f acc10 = {}, acc11 = {}, acc12 = {}, acc13 = {};

  const __bf16* brow = Wih + (size_t)(tileN + l15) * K1;

  for (int k0 = 0; k0 < K1; k0 += 32) {
    const int ka = k0 + g * 8;
    const int kb = k0 + 16 + g * 8;

    // ---- A fragments (two contiguous 8-half runs per lane per M-tile)
    const float* s1 = (ka < Dd) ? (rw0 + ka) : (mb0 + (ka - Dd));
    const float* s2 = (kb < Dd) ? (rw0 + kb) : (mb0 + (kb - Dd));
    const float* s3 = (ka < Dd) ? (rw1 + ka) : (mb1 + (ka - Dd));
    const float* s4 = (kb < Dd) ? (rw1 + kb) : (mb1 + (kb - Dd));
    float4 f0 = *(const float4*)(s1);
    float4 f1 = *(const float4*)(s1 + 4);
    float4 f2 = *(const float4*)(s2);
    float4 f3 = *(const float4*)(s2 + 4);
    float4 f4 = *(const float4*)(s3);
    float4 f5 = *(const float4*)(s3 + 4);
    float4 f6 = *(const float4*)(s4);
    float4 f7 = *(const float4*)(s4 + 4);
    v16bf a0, a1;
    a0[0]  = (__bf16)f0.x; a0[1]  = (__bf16)f0.y; a0[2]  = (__bf16)f0.z; a0[3]  = (__bf16)f0.w;
    a0[4]  = (__bf16)f1.x; a0[5]  = (__bf16)f1.y; a0[6]  = (__bf16)f1.z; a0[7]  = (__bf16)f1.w;
    a0[8]  = (__bf16)f2.x; a0[9]  = (__bf16)f2.y; a0[10] = (__bf16)f2.z; a0[11] = (__bf16)f2.w;
    a0[12] = (__bf16)f3.x; a0[13] = (__bf16)f3.y; a0[14] = (__bf16)f3.z; a0[15] = (__bf16)f3.w;
    a1[0]  = (__bf16)f4.x; a1[1]  = (__bf16)f4.y; a1[2]  = (__bf16)f4.z; a1[3]  = (__bf16)f4.w;
    a1[4]  = (__bf16)f5.x; a1[5]  = (__bf16)f5.y; a1[6]  = (__bf16)f5.z; a1[7]  = (__bf16)f5.w;
    a1[8]  = (__bf16)f6.x; a1[9]  = (__bf16)f6.y; a1[10] = (__bf16)f6.z; a1[11] = (__bf16)f6.w;
    a1[12] = (__bf16)f7.x; a1[13] = (__bf16)f7.y; a1[14] = (__bf16)f7.z; a1[15] = (__bf16)f7.w;

    // ---- B fragments: two contiguous 8-half runs per lane per n-tile
    Frag16 fb0, fb1, fb2, fb3;
    fb0.q[0] = *(const uint4*)(brow + ka); fb0.q[1] = *(const uint4*)(brow + kb);
    fb1.q[0] = *(const uint4*)(brow + (size_t)16 * K1 + ka); fb1.q[1] = *(const uint4*)(brow + (size_t)16 * K1 + kb);
    fb2.q[0] = *(const uint4*)(brow + (size_t)32 * K1 + ka); fb2.q[1] = *(const uint4*)(brow + (size_t)32 * K1 + kb);
    fb3.q[0] = *(const uint4*)(brow + (size_t)48 * K1 + ka); fb3.q[1] = *(const uint4*)(brow + (size_t)48 * K1 + kb);
    v16bf b0 = fb0.v, b1 = fb1.v, b2 = fb2.v, b3 = fb3.v;

    acc00 = __builtin_amdgcn_wmma_f32_16x16x32_bf16(false, a0, false, b0, (short)0, acc00, false, false);
    acc01 = __builtin_amdgcn_wmma_f32_16x16x32_bf16(false, a0, false, b1, (short)0, acc01, false, false);
    acc02 = __builtin_amdgcn_wmma_f32_16x16x32_bf16(false, a0, false, b2, (short)0, acc02, false, false);
    acc03 = __builtin_amdgcn_wmma_f32_16x16x32_bf16(false, a0, false, b3, (short)0, acc03, false, false);
    acc10 = __builtin_amdgcn_wmma_f32_16x16x32_bf16(false, a1, false, b0, (short)0, acc10, false, false);
    acc11 = __builtin_amdgcn_wmma_f32_16x16x32_bf16(false, a1, false, b1, (short)0, acc11, false, false);
    acc12 = __builtin_amdgcn_wmma_f32_16x16x32_bf16(false, a1, false, b2, (short)0, acc12, false, false);
    acc13 = __builtin_amdgcn_wmma_f32_16x16x32_bf16(false, a1, false, b3, (short)0, acc13, false, false);
  }

  // ---- epilogue: D layout row = r + 8*g, col = l15
  const float bi0 = b_ih[tileN      + l15];
  const float bi1 = b_ih[tileN + 16 + l15];
  const float bi2 = b_ih[tileN + 32 + l15];
  const float bi3 = b_ih[tileN + 48 + l15];
#pragma unroll
  for (int rr = 0; rr < 8; ++rr) {
    const int row0 = tileM + rr + 8 * g;
    float* o0 = gi + (size_t)row0 * N1 + tileN + l15;
    o0[0]  = acc00[rr] + bi0;
    o0[16] = acc01[rr] + bi1;
    o0[32] = acc02[rr] + bi2;
    o0[48] = acc03[rr] + bi3;
    float* o1 = o0 + (size_t)16 * N1;
    o1[0]  = acc10[rr] + bi0;
    o1[16] = acc11[rr] + bi1;
    o1[32] = acc12[rr] + bi2;
    o1[48] = acc13[rr] + bi3;
  }
}

// ------------------------------------------------------------------
// device-wide sense barrier
// ------------------------------------------------------------------
__device__ __forceinline__ void grid_barrier(unsigned* cnt, unsigned* gen, unsigned nwg) {
  __syncthreads();
  __threadfence();
  if (threadIdx.x == 0) {
    unsigned g = __hip_atomic_load(gen, __ATOMIC_ACQUIRE, __HIP_MEMORY_SCOPE_AGENT);
    unsigned old = __hip_atomic_fetch_add(cnt, 1u, __ATOMIC_ACQ_REL, __HIP_MEMORY_SCOPE_AGENT);
    if (old == nwg - 1u) {
      __hip_atomic_store(cnt, 0u, __ATOMIC_RELAXED, __HIP_MEMORY_SCOPE_AGENT);
      __hip_atomic_fetch_add(gen, 1u, __ATOMIC_ACQ_REL, __HIP_MEMORY_SCOPE_AGENT);
    } else {
      while (__hip_atomic_load(gen, __ATOMIC_ACQUIRE, __HIP_MEMORY_SCOPE_AGENT) == g) {
        __builtin_amdgcn_s_sleep(1);
      }
    }
  }
  __threadfence();
  __syncthreads();
}

// ------------------------------------------------------------------
// Phase 2: persistent GRU scan. 64 WGs x 128 threads; WG owns 16 dims.
// h (8x1024 bf16) staged into LDS via GLOBAL_LOAD_ASYNC_TO_LDS_B128;
// pad rows 8..15 of the LDS A-tile are zeroed once.
// Waves 0..2: gh tiles (r/z/n) via bf16 WMMA over K=1024; wave 3 prefetch.
// ------------------------------------------------------------------
__global__ __launch_bounds__(128) void gru_scan_kernel(
    const float* __restrict__ rwkv, const float* __restrict__ mamba,
    const float* __restrict__ hidden,
    const void* __restrict__ whh_bf_, const float* __restrict__ b_hh,
    const float* __restrict__ w_out, const float* __restrict__ b_out,
    const float* __restrict__ gi, void* hbuf_, float* __restrict__ logits,
    unsigned* __restrict__ bar,
    float* __restrict__ out_fused, float* __restrict__ out_weights,
    float* __restrict__ out_hlast) {
  const __bf16* Whh = (const __bf16*)whh_bf_;
  __bf16* hbuf = (__bf16*)hbuf_;  // [2][8][Dd] bf16

  __shared__ __align__(16) __bf16 hA[16 * Dd];  // 32 KB padded h (rows 8..15 zero)
  __shared__ float ghs[3][16][16];

  const int tid  = threadIdx.x;
  const int wave = tid >> 5;
  const int lane = tid & 31;
  const int l15  = lane & 15;
  const int g    = lane >> 4;
  const int d0   = blockIdx.x * 16;  // this WG's dim slice

  const int b  = tid >> 4;   // 0..7  (128 threads = 8 batches x 16 dims)
  const int dl = tid & 15;
  const int d  = d0 + dl;

  float hprev = hidden[(size_t)b * Dd + d];

  // zero pad rows 8..15 of the LDS A-tile once
  {
    uint4 z = make_uint4(0u, 0u, 0u, 0u);
    uint4* dst = (uint4*)hA;
#pragma unroll
    for (int i = 0; i < 8; ++i) dst[1024 + tid + 128 * i] = z;
  }

  const unsigned ldsBase = (unsigned)(uintptr_t)(void*)hA;

  for (int t = 0; t < Ll; ++t) {
    const int p = t & 1;

    // ---- stage h rows 0..7 (8x1024 bf16 = 16KB) via async DMA to LDS
    {
      const __bf16* src = hbuf + (size_t)p * Bb * Dd;
#pragma unroll
      for (int i = 0; i < 8; ++i) {
        const int e = tid + 128 * i;  // uint4 index 0..1023
        unsigned la = ldsBase + (unsigned)(e * 16);
        unsigned long long ga = (unsigned long long)(uintptr_t)(src + e * 8);
        asm volatile("global_load_async_to_lds_b128 %0, %1, off"
                     :: "v"(la), "v"(ga) : "memory");
      }
      asm volatile("s_wait_asynccnt 0" ::: "memory");
    }
    __syncthreads();

    if (wave < 3) {
      const int gate = wave;  // 0=r, 1=z, 2=n
      const __bf16* wrow = Whh + ((size_t)gate * Dd + d0 + l15) * Dd;
      v8f acc = {};
      for (int k0 = 0; k0 < Dd; k0 += 32) {
        Frag16 fa;
        fa.q[0] = *(const uint4*)(hA + l15 * Dd + k0 + g * 8);
        fa.q[1] = *(const uint4*)(hA + l15 * Dd + k0 + 16 + g * 8);
        Frag16 fbw;
        fbw.q[0] = *(const uint4*)(wrow + k0 + g * 8);
        fbw.q[1] = *(const uint4*)(wrow + k0 + 16 + g * 8);
        v16bf bf = fbw.v;
        acc = __builtin_amdgcn_wmma_f32_16x16x32_bf16(false, fa.v, false, bf, (short)0, acc, false, false);
      }
#pragma unroll
      for (int rr = 0; rr < 8; ++rr) ghs[gate][rr + 8 * g][l15] = acc[rr];
    } else {
      // wave 3: warm L2/L0 for the elementwise stage
      if (lane < Bb) {
        const float* gp = gi + ((size_t)lane * Ll + t) * N1 + d0;
        __builtin_prefetch(gp, 0, 1);
        __builtin_prefetch(gp + Dd, 0, 1);
        __builtin_prefetch(gp + 2 * Dd, 0, 1);
        __builtin_prefetch(rwkv  + ((size_t)lane * Ll + t) * Dd + d0, 0, 1);
        __builtin_prefetch(mamba + ((size_t)lane * Ll + t) * Dd + d0, 0, 1);
      }
    }
    __syncthreads();

    // ---- gates: thread = (batch b, dim d)
    const float* grow = gi + ((size_t)b * Ll + t) * N1;
    const float gir = grow[d];
    const float giz = grow[Dd + d];
    const float gin = grow[2 * Dd + d];
    const float ghr = ghs[0][b][dl] + b_hh[d];
    const float ghz = ghs[1][b][dl] + b_hh[Dd + d];
    const float ghn = ghs[2][b][dl] + b_hh[2 * Dd + d];
    const float r = sigmoidf_(gir + ghr);
    const float z = sigmoidf_(giz + ghz);
    const float n = tanhf(gin + r * ghn);
    const float hnew = (1.0f - z) * n + z * hprev;
    hprev = hnew;
    hbuf[(size_t)(1 - p) * Bb * Dd + (size_t)b * Dd + d] = (__bf16)hnew;

    // ---- partial logits: 16-lane reduction within each batch group
    float p0 = hnew * w_out[d];
    float p1 = hnew * w_out[Dd + d];
#pragma unroll
    for (int off = 8; off >= 1; off >>= 1) {
      p0 += __shfl_xor(p0, off, 32);
      p1 += __shfl_xor(p1, off, 32);
    }
    if (dl == 0) {
      atomicAdd(&logits[t * 16 + b * 2 + 0], p0);
      atomicAdd(&logits[t * 16 + b * 2 + 1], p1);
    }

    grid_barrier(bar, bar + 1, (unsigned)gridDim.x);

    // ---- post-barrier: softmax fusion weights + fused output
    const float l0 = logits[t * 16 + b * 2 + 0] + b_out[0];
    const float l1 = logits[t * 16 + b * 2 + 1] + b_out[1];
    const float mx = fmaxf(l0, l1);
    const float e0 = __expf(l0 - mx);
    const float e1 = __expf(l1 - mx);
    const float inv = 1.0f / (e0 + e1);
    const float w0 = e0 * inv, w1 = e1 * inv;
    const size_t xi = ((size_t)b * Ll + t) * Dd + d;
    out_fused[xi] = w0 * rwkv[xi] + w1 * mamba[xi];

    if (blockIdx.x == 0 && tid < 16) {
      const int bb = tid >> 1, jj = tid & 1;
      const float q0 = logits[t * 16 + bb * 2 + 0] + b_out[0];
      const float q1 = logits[t * 16 + bb * 2 + 1] + b_out[1];
      const float qm = fmaxf(q0, q1);
      const float x0 = __expf(q0 - qm);
      const float x1 = __expf(q1 - qm);
      out_weights[((size_t)bb * Ll + t) * 2 + jj] = (jj == 0 ? x0 : x1) / (x0 + x1);
    }
  }

  out_hlast[(size_t)b * Dd + d] = hprev;
}

// ------------------------------------------------------------------
extern "C" void kernel_launch(void* const* d_in, const int* in_sizes, int n_in,
                              void* d_out, int out_size, void* d_ws, size_t ws_size,
                              hipStream_t stream) {
  const float* rwkv   = (const float*)d_in[0];
  const float* mamba  = (const float*)d_in[1];
  const float* hidden = (const float*)d_in[2];
  const float* w_ih   = (const float*)d_in[3];
  const float* w_hh   = (const float*)d_in[4];
  const float* b_ih   = (const float*)d_in[5];
  const float* b_hh   = (const float*)d_in[6];
  const float* w_out  = (const float*)d_in[7];
  const float* b_out  = (const float*)d_in[8];

  char* ws = (char*)d_ws;
  size_t o = 0;
  void* wih_bf = ws + o;  o += (size_t)N1 * K1 * 2;          // 12 MB bf16 W_ih
  void* whh_bf = ws + o;  o += (size_t)N1 * Dd * 2;          // 6 MB  bf16 W_hh
  float* gi    = (float*)(ws + o); o += (size_t)M1 * N1 * 4; // 192 MB f32 gi
  void* hbuf   = ws + o;  o += (size_t)2 * Bb * Dd * 2;      // double-buffered h (bf16)
  float* logits = (float*)(ws + o); o += (size_t)Ll * 16 * 4;
  unsigned* bar = (unsigned*)(ws + o); o += 256;

  float* out_fused   = (float*)d_out;
  float* out_weights = out_fused + (size_t)Bb * Ll * Dd;
  float* out_hlast   = out_weights + (size_t)Bb * Ll * 2;

  init_ws_kernel<<<2048, 256, 0, stream>>>(w_ih, w_hh, hidden, wih_bf, whh_bf,
                                           hbuf, logits, bar);

  // 24576 wave-tiles (512 M32-tiles x 48 N64-tiles), 8 waves per block
  gemm_ih_kernel<<<3072, 256, 0, stream>>>(rwkv, mamba, wih_bf, b_ih, gi);

  gru_scan_kernel<<<SCAN_WGS, 128, 0, stream>>>(rwkv, mamba, hidden, whh_bf, b_hh,
                                                w_out, b_out, gi, hbuf, logits, bar,
                                                out_fused, out_weights, out_hlast);
}
